// ODEModel_14250701488510
// MI455X (gfx1250) — matrix operand order repaired and checked by reading, not compiled, not run on hardware
//
#include <hip/hip_runtime.h>

typedef __attribute__((ext_vector_type(16))) _Float16 v16h;
typedef __attribute__((ext_vector_type(8)))  _Float16 v8h;
typedef __attribute__((ext_vector_type(8)))  float    v8f;
typedef __attribute__((ext_vector_type(4)))  float    v4f;

// Hardware v_tanh_f32 on gfx1250 (probe-confirmed); software fallback otherwise.
__device__ __forceinline__ float tanh_fast(float x) {
#if __has_builtin(__builtin_amdgcn_tanhf)
  return __builtin_amdgcn_tanhf(x);
#else
  float e = __builtin_amdgcn_exp2f(2.8853900817779268f * x);
  return 1.0f - 2.0f * __builtin_amdgcn_rcpf(e + 1.0f);
#endif
}

// Single wave32 carries the whole sequential RK4 scan (latency-bound problem:
// ~0.7 GFLOP total, 1 MB output -> neither HBM nor matrix-peak matter; only the
// per-layer dependency chain does). Weights live in VGPRs in WMMA A-layout.
// The activation vector is broadcast into all 16 B columns, so every lane gets
// the full result in its own C registers. W3/b3 are replicated into rows 8-11
// of the padded 16x32 A tile so BOTH lane-halves receive k[0..3] in c2[0..3],
// letting every lane keep a private full copy of the RK4 state in registers
// (no LDS round-trip, no cross-lane selects, no divergence on the hot path).
__global__ __launch_bounds__(32) void ode_rk4_wmma_kernel(
    const float* __restrict__ s_grid, const float* __restrict__ y0,
    const float* __restrict__ W1, const float* __restrict__ b1,
    const float* __restrict__ W2, const float* __restrict__ b2,
    const float* __restrict__ W3, const float* __restrict__ b3,
    float* __restrict__ out, int nsteps)
{
  const int lane = threadIdx.x;   // 0..31 (wave32)
  const int kh   = lane >> 4;     // which K-half this lane holds (A/B layout)
  const int mrow = lane & 15;     // tile row (A) / column (B,C)

  __shared__ __align__(32) _Float16 actA[32];  // h1 in B-layout (index == K)
  __shared__ __align__(32) _Float16 actB[32];  // h2 in B-layout

  // ---- layer-1 weights: full f32, one hidden row per lane ----
  float w1r[5];
#pragma unroll
  for (int j = 0; j < 5; ++j) w1r[j] = W1[lane * 5 + j];
  const float b1r = b1[lane];

  // ---- W2 / W3 in 16-bit A-matrix 16x32 layout ----
  // element e (VGPR e/2, half e&1):  K = e + 16*kh
  v16h a2t0, a2t1, a3;
#pragma unroll
  for (int e = 0; e < 16; ++e) {
    const int k = e + 16 * kh;
    a2t0[e] = (_Float16)W2[(mrow)      * 32 + k];
    a2t1[e] = (_Float16)W2[(mrow + 16) * 32 + k];
    // W3 (4x32) in rows 0-3 AND replicated in rows 8-11 (zero elsewhere)
    float w3v = 0.0f;
    if (mrow < 4)                    w3v = W3[mrow * 32 + k];
    else if (mrow >= 8 && mrow < 12) w3v = W3[(mrow - 8) * 32 + k];
    a3[e] = (_Float16)w3v;
  }

  // ---- biases folded into C-initializers (C layout: VGPR r -> row r + 8*kh) ----
  v8f cb2t0, cb2t1, cb3;
#pragma unroll
  for (int r = 0; r < 8; ++r) {
    const int m = r + 8 * kh;
    cb2t0[r] = b2[m];
    cb2t1[r] = b2[m + 16];
    float b3v = 0.0f;
    if (m < 4)                  b3v = b3[m];
    else if (m >= 8 && m < 12)  b3v = b3[m - 8];
    cb3[r] = b3v;
  }

  // ---- RK4 state: full 4-vector kept redundantly in EVERY lane ----
  v4f y;
  y.x = y0[0]; y.y = y0[1]; y.z = y0[2]; y.w = y0[3];
  if (lane == 0) *(v4f*)out = y;

  float s_cur = s_grid[0];

  // One network evaluation: z (4-vector) + s from registers; returns k 4-vector
  // valid in every lane.
  auto feval = [&](v4f z, float s) -> v4f {
    // layer 1: f32 VALU, one hidden row per lane (keeps s at full precision)
    float a = b1r + w1r[0] * z.x + w1r[1] * z.y + w1r[2] * z.z +
              w1r[3] * z.w + w1r[4] * s;
    actA[lane] = (_Float16)tanh_fast(a);
    asm volatile("" ::: "memory");

    // layer 2: two WMMA 16x16x32 f16 tiles, h1 broadcast across all 16 columns
    v16h bv = *(const v16h*)(actA + 16 * kh);
    v8f c0 = __builtin_amdgcn_wmma_f32_16x16x32_f16(false, a2t0, false, bv,
                                                    (short)0, cb2t0, false, false);
    v8f c1 = __builtin_amdgcn_wmma_f32_16x16x32_f16(false, a2t1, false, bv,
                                                    (short)0, cb2t1, false, false);
    v8h p0, p1;
#pragma unroll
    for (int r = 0; r < 8; ++r) {
      p0[r] = (_Float16)tanh_fast(c0[r]);
      p1[r] = (_Float16)tanh_fast(c1[r]);
    }
    // lane 0 holds rows 0..7 / 16..23, lane 16 holds rows 8..15 / 24..31
    if (mrow == 0) {
      *(v8h*)(actB + 8 * kh)      = p0;
      *(v8h*)(actB + 16 + 8 * kh) = p1;
    }
    asm volatile("" ::: "memory");

    // layer 3: one WMMA; k[0..3] lands in c2[0..3] of EVERY lane
    // (rows 0-3 for lanes 0-15, replicated rows 8-11 for lanes 16-31)
    v16h bv2 = *(const v16h*)(actB + 16 * kh);
    v8f c2 = __builtin_amdgcn_wmma_f32_16x16x32_f16(false, a3, false, bv2,
                                                    (short)0, cb3, false, false);
    v4f kv;
    kv.x = c2[0]; kv.y = c2[1]; kv.z = c2[2]; kv.w = c2[3];
    return kv;
  };

  for (int t = 0; t < nsteps; ++t) {
    float s_nxt = s_grid[t + 1];   // uniform s_load; latency hidden by stage 1
    float hh    = s_nxt - s_cur;

    v4f k1 = feval(y, s_cur);
    v4f k2 = feval(y + 0.5f * hh * k1, s_cur);
    v4f k3 = feval(y + 0.5f * hh * k2, s_cur);
    v4f k4 = feval(y + hh * k3, s_cur);

    y = y + (hh * (1.0f / 6.0f)) * (k1 + 2.0f * k2 + 2.0f * k3 + k4);
    if (lane == 0) *(v4f*)(out + 4 * (t + 1)) = y;
    s_cur = s_nxt;
  }
}

extern "C" void kernel_launch(void* const* d_in, const int* in_sizes, int n_in,
                              void* d_out, int out_size, void* d_ws, size_t ws_size,
                              hipStream_t stream) {
  const float* s_grid = (const float*)d_in[0];
  const float* y0v    = (const float*)d_in[1];
  const float* W1     = (const float*)d_in[2];
  const float* b1     = (const float*)d_in[3];
  const float* W2     = (const float*)d_in[4];
  const float* b2     = (const float*)d_in[5];
  const float* W3     = (const float*)d_in[6];
  const float* b3     = (const float*)d_in[7];
  float* out = (float*)d_out;
  const int nsteps = in_sizes[0] - 1;   // N-1 RK4 steps

  hipLaunchKernelGGL(ode_rk4_wmma_kernel, dim3(1), dim3(32), 0, stream,
                     s_grid, y0v, W1, b1, W2, b2, W3, b3, out, nsteps);
}
